// GINGraphNet_ENZYMES_34832184770976
// MI455X (gfx1250) — matrix operand order repaired and checked by reading, not compiled, not run on hardware
//
#include <hip/hip_runtime.h>
#include <hip/hip_bf16.h>

typedef __attribute__((ext_vector_type(16))) _Float16 v16h;
typedef __attribute__((ext_vector_type(8)))  float    v8f;
typedef __attribute__((ext_vector_type(4)))  float    f4;

#define WAVES_PER_BLOCK 4

// ---------------- utility kernels ----------------

__global__ void zero_kernel(float* p, int n) {
    int i = blockIdx.x * blockDim.x + threadIdx.x;
    if (i < n) p[i] = 0.0f;
}

__global__ void copy_f4_kernel(const f4* __restrict__ src, f4* __restrict__ dst, long n4) {
    long i = (long)blockIdx.x * blockDim.x + threadIdx.x;
    if (i < n4) dst[i] = src[i];
}

// ---------------- edge scatter (GIN aggregation) ----------------
// one wave per edge; 32 lanes = 32 features (coalesced row); edge indices are
// wave-uniform -> readfirstlane so they become scalar (SMEM) loads.
__global__ void scatter32_kernel(const int* __restrict__ srcIdx, const int* __restrict__ dstIdx,
                                 const float* __restrict__ x, float* __restrict__ agg, long nEdges) {
    int wlocal = __builtin_amdgcn_readfirstlane((int)(threadIdx.x >> 5));
    long e = (long)blockIdx.x * (blockDim.x >> 5) + wlocal;
    if (e >= nEdges) return;
    int f = threadIdx.x & 31;
    long s = (long)srcIdx[e];
    long d = (long)dstIdx[e];
    atomicAdd(&agg[d * 32 + f], x[s * 32 + f]);
}

// one wave per 2 edges; lanes 0-15 = edge0 features, lanes 16-31 = edge1 features
__global__ void scatter16_kernel(const int* __restrict__ srcIdx, const int* __restrict__ dstIdx,
                                 const float* __restrict__ x, float* __restrict__ agg, long nEdges) {
    int wlocal = __builtin_amdgcn_readfirstlane((int)(threadIdx.x >> 5));
    long ebase = ((long)blockIdx.x * (blockDim.x >> 5) + wlocal) * 2;
    if (ebase >= nEdges) return;
    int lane = threadIdx.x & 31;
    int f  = lane & 15;
    int hi = lane >> 4;
    long e1 = (ebase + 1 < nEdges) ? ebase + 1 : ebase;   // clamp (nEdges is even anyway)
    int s0 = srcIdx[ebase], d0 = dstIdx[ebase];
    int s1 = srcIdx[e1],    d1 = dstIdx[e1];
    long s = hi ? (long)s1 : (long)s0;
    long d = hi ? (long)d1 : (long)d0;
    atomicAdd(&agg[d * 16 + f], x[s * 16 + f]);
}

// ---------------- WMMA fragment index helper ----------------
// B (32x16 f16): lane = column (n = lane&15); elements i<8 hold K 8*hi+i,
// elements i>=8 hold K 16+8*hi+(i-8) (same order as A)
__device__ __forceinline__ int b_kmap(int i, int hi) {
    return (i < 8) ? (i + (hi << 3)) : (16 + (hi << 3) + (i - 8));
}

// ---------------- MLP of GIN layer 1: [N,32] -> relu(.@W1a+b1a) -> relu(.@W1b+b1b) -> [N,16]
__global__ void mlp1_kernel(const float* __restrict__ h0,
                            const float* __restrict__ W1a, const float* __restrict__ b1a,
                            const float* __restrict__ W1b, const float* __restrict__ b1b,
                            float* __restrict__ h1, int nNodes) {
    __shared__ float Wa_s[32][17];   // full 32x16 W1a tile, no zero padding
    __shared__ float Wb_s[32][17];   // rows >= 16 zero
    __shared__ float stage[WAVES_PER_BLOCK][16][17];

    for (int idx = threadIdx.x; idx < 512; idx += blockDim.x) {
        int k = idx >> 4, n = idx & 15;
        Wa_s[k][n] = W1a[k * 16 + n];                       // [32,16] full
        Wb_s[k][n] = (k < 16) ? W1b[k * 16 + n] : 0.0f;     // [16,16] K-padded
    }
    __syncthreads();

    int lane = threadIdx.x & 31;
    int wave = threadIdx.x >> 5;
    int tile = blockIdx.x * WAVES_PER_BLOCK + wave;
    long base = (long)tile * 16;
    int m  = lane & 15;
    int hi = lane >> 4;
    int n  = m;
    long node  = base + m;
    long nodeC = node < nNodes ? node : (long)nNodes - 1;   // clamp; extra rows never stored

    // A fragment: lane row m; K runs [8*hi .. 8*hi+7] and [16+8*hi .. 16+8*hi+7]
    const float* row = h0 + nodeC * 32 + (hi << 3);
    f4 q0 = *(const f4*)(row);
    f4 q1 = *(const f4*)(row + 4);
    f4 q2 = *(const f4*)(row + 16);
    f4 q3 = *(const f4*)(row + 20);
    v16h a, b;
    a[0]=(_Float16)q0.x; a[1]=(_Float16)q0.y; a[2]=(_Float16)q0.z; a[3]=(_Float16)q0.w;
    a[4]=(_Float16)q1.x; a[5]=(_Float16)q1.y; a[6]=(_Float16)q1.z; a[7]=(_Float16)q1.w;
    a[8]=(_Float16)q2.x; a[9]=(_Float16)q2.y; a[10]=(_Float16)q2.z; a[11]=(_Float16)q2.w;
    a[12]=(_Float16)q3.x; a[13]=(_Float16)q3.y; a[14]=(_Float16)q3.z; a[15]=(_Float16)q3.w;

    #pragma unroll
    for (int i = 0; i < 16; ++i) b[i] = (_Float16)Wa_s[b_kmap(i, hi)][n];

    v8f c = {};
    c = __builtin_amdgcn_wmma_f32_16x16x32_f16(false, a, false, b, (short)0, c, false, false);

    float bias = b1a[n];
    #pragma unroll
    for (int r = 0; r < 8; ++r) {
        int mm = r + (hi << 3);
        float t = c[r] + bias;
        stage[wave][mm][n] = t > 0.0f ? t : 0.0f;
    }
    __syncthreads();

    // second linear: K=16; elements i<8 -> k = i + 8*hi (always < 16); i>=8 -> k>=16 -> const zero
    #pragma unroll
    for (int i = 0; i < 8; ++i) a[i] = (_Float16)stage[wave][m][i + (hi << 3)];
    #pragma unroll
    for (int i = 8; i < 16; ++i) a[i] = (_Float16)0.0f;
    #pragma unroll
    for (int i = 0; i < 16; ++i) b[i] = (_Float16)Wb_s[b_kmap(i, hi)][n];

    v8f c2 = {};
    c2 = __builtin_amdgcn_wmma_f32_16x16x32_f16(false, a, false, b, (short)0, c2, false, false);

    float bias2 = b1b[n];
    #pragma unroll
    for (int r = 0; r < 8; ++r) {
        int mm = r + (hi << 3);
        long nd = base + mm;
        if (nd < nNodes) {
            float t = c2[r] + bias2;
            h1[nd * 16 + n] = t > 0.0f ? t : 0.0f;
        }
    }
}

// ---------------- MLP of GIN layer 2 fused with mean-pool scatter:
// [N,16] -> relu(.@W2a+b2a) [N,8] -> relu(.@W2b+b2b) [N,8] -> atomic add into sums[batch,8]
__global__ void mlp2_pool_kernel(const float* __restrict__ h2,
                                 const float* __restrict__ W2a, const float* __restrict__ b2a,
                                 const float* __restrict__ W2b, const float* __restrict__ b2b,
                                 const int* __restrict__ batch, float* __restrict__ sums,
                                 int nNodes) {
    __shared__ float Wa_s[32][17];   // zero-padded [16,8] -> 32x16
    __shared__ float Wb_s[32][17];   // zero-padded [8,8]  -> 32x16
    __shared__ float ba_s[16];       // zero-padded biases
    __shared__ float bb_s[16];
    __shared__ float stage[WAVES_PER_BLOCK][16][17];

    for (int idx = threadIdx.x; idx < 512; idx += blockDim.x) {
        int k = idx >> 4, n = idx & 15;
        Wa_s[k][n] = (k < 16 && n < 8) ? W2a[k * 8 + n] : 0.0f;
        Wb_s[k][n] = (k < 8  && n < 8) ? W2b[k * 8 + n] : 0.0f;
    }
    if (threadIdx.x < 16) {
        ba_s[threadIdx.x] = (threadIdx.x < 8) ? b2a[threadIdx.x] : 0.0f;
        bb_s[threadIdx.x] = (threadIdx.x < 8) ? b2b[threadIdx.x] : 0.0f;
    }
    __syncthreads();

    int lane = threadIdx.x & 31;
    int wave = threadIdx.x >> 5;
    int tile = blockIdx.x * WAVES_PER_BLOCK + wave;
    long base = (long)tile * 16;
    int m  = lane & 15;
    int hi = lane >> 4;
    int n  = m;
    long node  = base + m;
    long nodeC = node < nNodes ? node : (long)nNodes - 1;

    // A fragment: K=16 total; i<8 -> k = i + 8*hi from h2 row; i>=8 -> k>=16 -> zero
    const float* row = h2 + nodeC * 16 + (hi << 3);
    f4 q0 = *(const f4*)(row);
    f4 q1 = *(const f4*)(row + 4);
    v16h a, b;
    a[0]=(_Float16)q0.x; a[1]=(_Float16)q0.y; a[2]=(_Float16)q0.z; a[3]=(_Float16)q0.w;
    a[4]=(_Float16)q1.x; a[5]=(_Float16)q1.y; a[6]=(_Float16)q1.z; a[7]=(_Float16)q1.w;
    #pragma unroll
    for (int i = 8; i < 16; ++i) a[i] = (_Float16)0.0f;
    #pragma unroll
    for (int i = 0; i < 16; ++i) b[i] = (_Float16)Wa_s[b_kmap(i, hi)][n];

    v8f c = {};
    c = __builtin_amdgcn_wmma_f32_16x16x32_f16(false, a, false, b, (short)0, c, false, false);

    // columns n>=8 have zero weights and zero bias -> t stays 0 there
    float bias = ba_s[n];
    #pragma unroll
    for (int r = 0; r < 8; ++r) {
        int mm = r + (hi << 3);
        float t = c[r] + bias;
        stage[wave][mm][n] = t > 0.0f ? t : 0.0f;
    }
    __syncthreads();

    // second linear: K=8 (Wb_s rows 8..31 zero; stage cols 8..15 zero)
    #pragma unroll
    for (int i = 0; i < 8; ++i) a[i] = (_Float16)stage[wave][m][i + (hi << 3)];
    #pragma unroll
    for (int i = 8; i < 16; ++i) a[i] = (_Float16)0.0f;
    #pragma unroll
    for (int i = 0; i < 16; ++i) b[i] = (_Float16)Wb_s[b_kmap(i, hi)][n];

    v8f c2 = {};
    c2 = __builtin_amdgcn_wmma_f32_16x16x32_f16(false, a, false, b, (short)0, c2, false, false);

    if (n < 8) {
        float bias2 = bb_s[n];
        #pragma unroll
        for (int r = 0; r < 8; ++r) {
            int mm = r + (hi << 3);
            long nd = base + mm;
            if (nd < nNodes) {
                float t = c2[r] + bias2;
                t = t > 0.0f ? t : 0.0f;
                int g = batch[nd];
                atomicAdd(&sums[(long)g * 8 + n], t);
            }
        }
    }
}

// ---------------- per-graph node counts ----------------
__global__ void counts_kernel(const int* __restrict__ batch, float* __restrict__ counts, int nNodes) {
    int i = blockIdx.x * blockDim.x + threadIdx.x;
    if (i < nNodes) atomicAdd(&counts[batch[i]], 1.0f);
}

// ---------------- final classifier + log_softmax ----------------
__global__ void final_kernel(const float* __restrict__ sums, const float* __restrict__ counts,
                             const float* __restrict__ Wfc, const float* __restrict__ bfc,
                             float* __restrict__ out, int nGraphs) {
    int g = blockIdx.x * blockDim.x + threadIdx.x;
    if (g >= nGraphs) return;
    float cnt = counts[g];
    cnt = cnt < 1.0f ? 1.0f : cnt;
    float p[8];
    #pragma unroll
    for (int k = 0; k < 8; ++k) p[k] = sums[(long)g * 8 + k] / cnt;
    float l[6];
    #pragma unroll
    for (int c = 0; c < 6; ++c) {
        float acc = bfc[c];
        #pragma unroll
        for (int k = 0; k < 8; ++k) acc += p[k] * Wfc[k * 6 + c];  // Wfc [8,6]
        l[c] = acc;
    }
    float mx = l[0];
    #pragma unroll
    for (int c = 1; c < 6; ++c) mx = l[c] > mx ? l[c] : mx;
    float s = 0.0f;
    #pragma unroll
    for (int c = 0; c < 6; ++c) s += __expf(l[c] - mx);
    float lse = __logf(s);
    #pragma unroll
    for (int c = 0; c < 6; ++c) out[(long)g * 6 + c] = l[c] - mx - lse;
}

extern "C" void kernel_launch(void* const* d_in, const int* in_sizes, int n_in,
                              void* d_out, int out_size, void* d_ws, size_t ws_size,
                              hipStream_t stream) {
    const float* x    = (const float*)d_in[0];
    const int*   edge = (const int*)d_in[1];
    const int*   batch= (const int*)d_in[2];
    const float* W1a  = (const float*)d_in[3];
    const float* b1a  = (const float*)d_in[4];
    const float* W1b  = (const float*)d_in[5];
    const float* b1b  = (const float*)d_in[6];
    const float* W2a  = (const float*)d_in[7];
    const float* b2a  = (const float*)d_in[8];
    const float* W2b  = (const float*)d_in[9];
    const float* b2b  = (const float*)d_in[10];
    const float* Wfc  = (const float*)d_in[11];
    const float* bfc  = (const float*)d_in[12];

    int  nNodes  = in_sizes[0] / 32;
    long nEdges  = (long)in_sizes[1] / 2;
    int  nGraphs = out_size / 6;

    const int* srcIdx = edge;            // edge_index[0]
    const int* dstIdx = edge + nEdges;   // edge_index[1]

    // workspace layout
    float* agg1   = (float*)d_ws;                       // [nNodes, 32]
    float* h1     = agg1 + (size_t)nNodes * 32;         // [nNodes, 16]
    float* agg2   = h1   + (size_t)nNodes * 16;         // [nNodes, 16]
    float* sums   = agg2 + (size_t)nNodes * 16;         // [nGraphs, 8]
    float* counts = sums + (size_t)nGraphs * 8;         // [nGraphs]

    // 1. zero sums + counts (accumulated via atomics each call)
    {
        int n = nGraphs * 9;
        zero_kernel<<<(n + 255) / 256, 256, 0, stream>>>(sums, n);
    }
    counts_kernel<<<(nNodes + 255) / 256, 256, 0, stream>>>(batch, counts, nNodes);

    // 2. agg1 = x  (so after scatter: agg1 = x + sum of neighbors)
    {
        long n4 = (long)nNodes * 8;
        copy_f4_kernel<<<(unsigned)((n4 + 255) / 256), 256, 0, stream>>>((const f4*)x, (f4*)agg1, n4);
    }
    // 3. scatter layer 1: one wave per edge, 8 waves per 256-thread block
    {
        long nWaveBlocks = (nEdges + 7) / 8;
        scatter32_kernel<<<(unsigned)nWaveBlocks, 256, 0, stream>>>(srcIdx, dstIdx, x, agg1, nEdges);
    }
    // 4. MLP1 (WMMA): agg1 -> h1
    {
        int nTiles = (nNodes + 15) / 16;
        int blocks = (nTiles + WAVES_PER_BLOCK - 1) / WAVES_PER_BLOCK;
        mlp1_kernel<<<blocks, WAVES_PER_BLOCK * 32, 0, stream>>>(agg1, W1a, b1a, W1b, b1b, h1, nNodes);
    }
    // 5. agg2 = h1
    {
        long n4 = (long)nNodes * 4;
        copy_f4_kernel<<<(unsigned)((n4 + 255) / 256), 256, 0, stream>>>((const f4*)h1, (f4*)agg2, n4);
    }
    // 6. scatter layer 2: 2 edges per wave
    {
        long nWaves = (nEdges + 1) / 2;
        long blocks = (nWaves + 7) / 8;
        scatter16_kernel<<<(unsigned)blocks, 256, 0, stream>>>(srcIdx, dstIdx, h1, agg2, nEdges);
    }
    // 7. MLP2 (WMMA) fused with mean-pool numerator scatter
    {
        int nTiles = (nNodes + 15) / 16;
        int blocks = (nTiles + WAVES_PER_BLOCK - 1) / WAVES_PER_BLOCK;
        mlp2_pool_kernel<<<blocks, WAVES_PER_BLOCK * 32, 0, stream>>>(agg2, W2a, b2a, W2b, b2b, batch, sums, nNodes);
    }
    // 8. classifier + log_softmax
    final_kernel<<<(nGraphs + 63) / 64, 64, 0, stream>>>(sums, counts, Wfc, bfc, (float*)d_out, nGraphs);
}
